// MultiHeadedAttention_85864986182332
// MI455X (gfx1250) — matrix operand order repaired and checked by reading, not compiled, not run on hardware
//
#include <hip/hip_runtime.h>

#define B_   8
#define R_   64
#define C_   256
#define NH_  8
#define N_   4096
#define HD_  32
#define NHEAD 64   // B_*NH_

typedef __attribute__((ext_vector_type(16))) _Float16 v16h;
typedef __attribute__((ext_vector_type(8)))  float    v8f;

// low 32 bits of a generic pointer to __shared__ = wave-relative LDS byte addr
__device__ __forceinline__ unsigned lds_addr_of(const void* p) {
    return (unsigned)(unsigned long long)(uintptr_t)p;
}

// ---------------------------------------------------------------------------
// Kernel 1: focused kernelization of q,k + head-split f16 copies.
//   qh : [head][n][d]  (row-major, A-operand of q@kv)
//   kT : [head][d][n]  (transposed, A-operand of k^T v and ksum rows)
//   vT : [head][d][n]  (transposed, B-operand of k^T v)
// One 256-thread block per (b,n) row; LDS tree reduction for the row norms.
// ---------------------------------------------------------------------------
__global__ void fla_prep(const float* __restrict__ qkv,
                         const float* __restrict__ pos,
                         const float* __restrict__ scale,
                         _Float16* __restrict__ qh,
                         _Float16* __restrict__ kT,
                         _Float16* __restrict__ vT) {
    const int c  = threadIdx.x;           // channel 0..255
    const int bn = blockIdx.x;            // 0..B*N-1
    const int b  = bn >> 12;
    const int n  = bn & (N_ - 1);
    const long base = (long)bn * C_ + c;

    float q = qkv[base];
    float k = qkv[(long)B_ * N_ * C_ + base] + pos[n * C_ + c];
    float v = qkv[2L * B_ * N_ * C_ + base];

    float sc = log1pf(expf(scale[c]));            // softplus
    float qr = (fmaxf(q, 0.f) + 1e-6f) / sc;
    float kr = (fmaxf(k, 0.f) + 1e-6f) / sc;
    float q3 = qr * qr * qr;
    float k3 = kr * kr * kr;

    __shared__ float4 red[256];
    red[c] = make_float4(qr * qr, q3 * q3, kr * kr, k3 * k3);
    __syncthreads();
    for (int s = 128; s > 0; s >>= 1) {
        if (c < s) {
            float4 o = red[c + s];
            red[c].x += o.x; red[c].y += o.y; red[c].z += o.z; red[c].w += o.w;
        }
        __syncthreads();
    }
    float4 t = red[0];
    float qf = q3 * rsqrtf(t.y) * sqrtf(t.x);     // q^3 / ||q^3|| * ||q||
    float kf = k3 * rsqrtf(t.w) * sqrtf(t.z);

    const int head = b * NH_ + (c >> 5);
    const int d    = c & 31;
    qh[((long)head * N_ + n) * HD_ + d]  = (_Float16)qf;
    kT[((long)head * HD_ + d) * N_ + n]  = (_Float16)kf;
    vT[((long)head * HD_ + d) * N_ + n]  = (_Float16)v;
}

// ---------------------------------------------------------------------------
// Kernel 2: k_sum[head][d] = sum_n k'[head][d][n]  (contiguous rows of kT)
// ---------------------------------------------------------------------------
__global__ void fla_ksum(const _Float16* __restrict__ kT,
                         float* __restrict__ ksum) {
    const int head = blockIdx.x;
    const int d    = threadIdx.x;                 // 32 threads
    const _Float16* p = kT + ((long)head * HD_ + d) * N_;
    float s = 0.f;
    for (int n = 0; n < N_; ++n) s += (float)p[n];
    ksum[head * HD_ + d] = s;
}

// ---------------------------------------------------------------------------
// Kernel 3: kv[head] = k'^T v  (32x32, K=4096) via WMMA f16->f32.
// 1 block / head, 4 waves = 4 output tiles. Double-buffered LDS staging
// through GLOBAL_LOAD_ASYNC_TO_LDS_B128 (ASYNCcnt), so the DMA of chunk
// i+1 overlaps the WMMA of chunk i with no VGPR bounce.
// ---------------------------------------------------------------------------
__global__ void fla_kv(const _Float16* __restrict__ kT,
                       const _Float16* __restrict__ vT,
                       _Float16* __restrict__ kvT) {
    const int head = blockIdx.x;
    const int tid  = threadIdx.x;                 // 128
    const int lane = tid & 31;
    const int wave = tid >> 5;
    const int mtile = wave >> 1;                  // c-tile of kv
    const int ntile = wave & 1;                   // d-tile of kv

    // [buf][k=0 / v=1][32 channels x 32 K] halves; 8 KB total
    __shared__ _Float16 stage[2][2][32 * 32];

    const int ldrow  = tid >> 2;                  // 0..31 channel row
    const int ldpart = tid & 3;                   // 16B part within 64B row
    const _Float16* ksrc = kT + ((long)head * HD_ + ldrow) * N_ + ldpart * 8;
    const _Float16* vsrc = vT + ((long)head * HD_ + ldrow) * N_ + ldpart * 8;

    const unsigned ldsK0 = lds_addr_of(&stage[0][0][0]) + (unsigned)tid * 16u;
    const unsigned ldsV0 = lds_addr_of(&stage[0][1][0]) + (unsigned)tid * 16u;
    const unsigned bufstride = (unsigned)sizeof(stage[0]);     // 4096 B

    // prologue: async DMA chunk 0 into buffer 0
    asm volatile("global_load_async_to_lds_b128 %0, %1, off"
                 :: "v"(ldsK0), "v"((unsigned long long)(uintptr_t)ksrc)
                 : "memory");
    asm volatile("global_load_async_to_lds_b128 %0, %1, off"
                 :: "v"(ldsV0), "v"((unsigned long long)(uintptr_t)vsrc)
                 : "memory");

    v8f acc = {0.f, 0.f, 0.f, 0.f, 0.f, 0.f, 0.f, 0.f};

    const int row = lane & 15;
    const int kb8 = (lane >> 4) << 3;             // A: K interleave by lane half
    const int kB0 = (lane >> 4) << 4;             // unused; B loads use the kb8 interleave like A
    const int cA  = mtile * 16 + row;             // kv row (channel c)
    const int dB  = ntile * 16 + row;             // kv col (channel d)

    for (int nk = 0; nk < N_; nk += 32) {
        const int buf = (nk >> 5) & 1;
        if (nk + 32 < N_) {                       // issue DMA for next chunk
            const unsigned nb = (unsigned)(buf ^ 1) * bufstride;
            asm volatile("global_load_async_to_lds_b128 %0, %1, off"
                         :: "v"(ldsK0 + nb),
                            "v"((unsigned long long)(uintptr_t)(ksrc + nk + 32))
                         : "memory");
            asm volatile("global_load_async_to_lds_b128 %0, %1, off"
                         :: "v"(ldsV0 + nb),
                            "v"((unsigned long long)(uintptr_t)(vsrc + nk + 32))
                         : "memory");
            asm volatile("s_wait_asynccnt 0x2" ::: "memory");   // chunk i landed
        } else {
            asm volatile("s_wait_asynccnt 0x0" ::: "memory");   // last chunk landed
        }
        __syncthreads();                          // all waves' DMA visible

        const _Float16* sk = stage[buf][0];
        const _Float16* sv = stage[buf][1];
        v16h a, bm;
        // A(c, K): halves 0..7 <- K = kb8..kb8+7 ; halves 8..15 <- K = 16+kb8..
        *(int4*)&a       = *(const int4*)&sk[cA * 32 + kb8];
        *((int4*)&a + 1) = *(const int4*)&sk[cA * 32 + 16 + kb8];
        // B(K, d): halves 0..7 <- K = kb8..kb8+7 ; halves 8..15 <- K = 16+kb8..
        *(int4*)&bm       = *(const int4*)&sv[dB * 32 + kb8];
        *((int4*)&bm + 1) = *(const int4*)&sv[dB * 32 + 16 + kb8];

        acc = __builtin_amdgcn_wmma_f32_16x16x32_f16(
            false, a, false, bm, (short)0, acc, false, false);
        __syncthreads();                          // reads done before buf reuse
    }

    _Float16* o = kvT + head * HD_ * HD_;         // [d][c]
    const int mofs = (lane >> 4) << 3;            // D layout: M = r + 8*(lane>=16)
#pragma unroll
    for (int r = 0; r < 8; ++r) {
        const int cc = mtile * 16 + mofs + r;
        o[dB * HD_ + cc] = (_Float16)acc[r];      // transposed store
    }
}

// ---------------------------------------------------------------------------
// Kernel 4: x = (q @ kv) * z, written to d_out in (B,N,C) layout.
// kv tile staged by one async-to-LDS DMA per thread (overlapped with the
// z computation); each wave owns 16 rows, K=32 in a single WMMA.
// ---------------------------------------------------------------------------
__global__ void fla_attn(const _Float16* __restrict__ qh,
                         const _Float16* __restrict__ kvT,
                         const float* __restrict__ ksum,
                         float* __restrict__ out) {
    const int head = blockIdx.x;
    const int tid  = threadIdx.x;                 // 128
    const int lane = tid & 31;
    const int wave = tid >> 5;

    __shared__ _Float16 skv[HD_ * HD_];           // kvT tile: [d][c]
    __shared__ float zsh[4][16];

    // async DMA the 2KB kv tile into LDS (16B per thread)
    const unsigned skv_lds = lds_addr_of(&skv[0]) + (unsigned)tid * 16u;
    const _Float16* kvsrc = kvT + head * HD_ * HD_ + tid * 8;
    asm volatile("global_load_async_to_lds_b128 %0, %1, off"
                 :: "v"(skv_lds), "v"((unsigned long long)(uintptr_t)kvsrc)
                 : "memory");

    const int rowbase = blockIdx.y * 64 + wave * 16;
    const _Float16* qb = qh + (long)head * N_ * HD_;

    if (lane < 16) {                              // z for this wave's 16 rows
        const _Float16* qr = qb + (long)(rowbase + lane) * HD_;
        const float* ks = ksum + head * HD_;
        float dot = 0.f;
#pragma unroll
        for (int d = 0; d < HD_; ++d) dot += (float)qr[d] * ks[d];
        zsh[wave][lane] = 1.f / (dot + 1e-6f);
    }
    asm volatile("s_wait_asynccnt 0x0" ::: "memory");
    __syncthreads();

    const int row = lane & 15;
    const int kb8 = (lane >> 4) << 3;
    const int kB0 = (lane >> 4) << 4;
    const _Float16* qr = qb + (long)(rowbase + row) * HD_;

    v16h a;                                       // A(n, c) row-major fragment
    *(int4*)&a       = *(const int4*)(qr + kb8);
    *((int4*)&a + 1) = *(const int4*)(qr + 16 + kb8);

    const int bb = head >> 3, hh = head & 7;
    const int mofs = (lane >> 4) << 3;

#pragma unroll
    for (int ntile = 0; ntile < 2; ++ntile) {
        const int d = ntile * 16 + row;
        v16h bm;                                  // B(c, d) from kvT[d][c]
        *(int4*)&bm       = *(const int4*)&skv[d * HD_ + kb8];
        *((int4*)&bm + 1) = *(const int4*)&skv[d * HD_ + 16 + kb8];

        v8f acc = {0.f, 0.f, 0.f, 0.f, 0.f, 0.f, 0.f, 0.f};
        acc = __builtin_amdgcn_wmma_f32_16x16x32_f16(
            false, a, false, bm, (short)0, acc, false, false);
#pragma unroll
        for (int r = 0; r < 8; ++r) {
            const int M = mofs + r;
            const int n = rowbase + M;
            out[((long)(bb * N_ + n)) * C_ + hh * HD_ + d] = zsh[wave][M] * acc[r];
        }
    }
}

// ---------------------------------------------------------------------------
// Kernel 5: fused depthwise convs (LePE 3x3 over C + 5x5 per head channel),
// accumulated into d_out. Both convs read the same v neighborhood.
// ---------------------------------------------------------------------------
__global__ void fla_conv(const float* __restrict__ qkv,
                         const float* __restrict__ w_v,
                         const float* __restrict__ b_v,
                         const float* __restrict__ w_dwc,
                         const float* __restrict__ b_dwc,
                         float* __restrict__ out) {
    const int c  = threadIdx.x;                   // channel
    const int bn = blockIdx.x;
    const int b  = bn >> 12;
    const int n  = bn & (N_ - 1);
    const int y  = n >> 6, x = n & 63;
    const int hd = c & 31;

    const float* v = qkv + 2L * B_ * N_ * C_ + (long)b * N_ * C_;

    float acc = b_v[c] + b_dwc[hd];
#pragma unroll
    for (int dy = -2; dy <= 2; ++dy) {
        const int yy = y + dy;
        if (yy < 0 || yy >= R_) continue;
#pragma unroll
        for (int dx = -2; dx <= 2; ++dx) {
            const int xx = x + dx;
            if (xx < 0 || xx >= R_) continue;
            const float vv = v[(long)(yy * R_ + xx) * C_ + c];
            acc += vv * w_dwc[hd * 25 + (dy + 2) * 5 + (dx + 2)];
            if (dy >= -1 && dy <= 1 && dx >= -1 && dx <= 1)
                acc += vv * w_v[c * 9 + (dy + 1) * 3 + (dx + 1)];
        }
    }
    out[(long)bn * C_ + c] += acc;
}

// ---------------------------------------------------------------------------
extern "C" void kernel_launch(void* const* d_in, const int* in_sizes, int n_in,
                              void* d_out, int out_size, void* d_ws, size_t ws_size,
                              hipStream_t stream) {
    const float* qkv   = (const float*)d_in[0];
    const float* pos   = (const float*)d_in[1];
    const float* scale = (const float*)d_in[2];
    const float* w_v   = (const float*)d_in[3];
    const float* b_v   = (const float*)d_in[4];
    const float* w_dwc = (const float*)d_in[5];
    const float* b_dwc = (const float*)d_in[6];
    float* out = (float*)d_out;

    char* ws = (char*)d_ws;
    _Float16* qh  = (_Float16*)(ws);                       // 16 MB
    _Float16* kT  = (_Float16*)(ws + 16777216);            // 16 MB
    _Float16* vT  = (_Float16*)(ws + 33554432);            // 16 MB
    _Float16* kvT = (_Float16*)(ws + 50331648);            // 128 KB
    float*    ksm = (float*)   (ws + 50462720);            // 8 KB

    fla_prep<<<dim3(B_ * N_), dim3(256), 0, stream>>>(qkv, pos, scale, qh, kT, vT);
    fla_ksum<<<dim3(NHEAD), dim3(32), 0, stream>>>(kT, ksm);
    fla_kv  <<<dim3(NHEAD), dim3(128), 0, stream>>>(kT, vT, kvT);
    fla_attn<<<dim3(NHEAD, N_ / 64), dim3(128), 0, stream>>>(qh, kvT, ksm, out);
    fla_conv<<<dim3(B_ * N_), dim3(256), 0, stream>>>(qkv, w_v, b_v, w_dwc, b_dwc, out);
}
